// SparseHilbertSSM_47115791237290
// MI455X (gfx1250) — compile-verified, matrix operand order checked by reading, not run on hardware
//
#include <hip/hip_runtime.h>
#include <hip/hip_bf16.h>
#include <math.h>

// ---------------------------------------------------------------------------
// SparseHilbertSSM forward for MI455X (gfx1250, wave32, WMMA).
// All dense GEMMs + SSD chunk einsums use v_wmma_f32_16x16x32_f16 (f16 in,
// f32 accum).  Elementwise math stays f32.  Entire working set (~106MB)
// is L2-resident (192MB), so the pipeline is launch/LDS/issue bound; GEMM
// uses 32x64 block tiles (2 WMMAs per wave per stage) + async global->LDS
// staging (ASYNCcnt) via the gfx1250 builtin.
// ---------------------------------------------------------------------------

typedef _Float16 h16;
typedef __attribute__((ext_vector_type(16))) h16  v16h;
typedef __attribute__((ext_vector_type(8)))  float v8f;

#define B_      2
#define L_      2048
#define DM_     256
#define DI_     512
#define NH_     8
#define CONV_   640
#define NPAD_   1216          // 1160 padded to a multiple of 64
#define ND_     6
#define NL_     2
#define NTOK_   (B_*L_)       // 4096
#define NCLS_   11

#if __has_builtin(__builtin_amdgcn_global_load_async_to_lds_b128)
#define HAVE_ASYNC_LDS 1
#endif

#ifdef HAVE_ASYNC_LDS
// Exact param type from the compiler diagnostic:
//   '__attribute__((__vector_size__(4*sizeof(int)))) int __device__ *'
typedef int v4i_ __attribute__((__vector_size__(16)));
typedef __attribute__((address_space(1))) v4i_* gv4p;   // global (AS1) int4*
typedef __attribute__((address_space(3))) v4i_* lv4p;   // LDS    (AS3) int4*
__device__ __forceinline__ void wait_async0(){
#if __has_builtin(__builtin_amdgcn_s_wait_asynccnt)
  __builtin_amdgcn_s_wait_asynccnt(0);
#else
  asm volatile("s_wait_asynccnt 0x0" ::: "memory");
#endif
}
#endif

__device__ __forceinline__ float geluf(float x){ return 0.5f*x*(1.0f+erff(x*0.70710678118654752f)); }
__device__ __forceinline__ float siluf(float x){ return x/(1.0f+__expf(-x)); }
__device__ __forceinline__ float softplusf(float x){ return (x>20.f)?x:log1pf(__expf(x)); }
// f16 A/B fragment element mapping (ISA 7.12.2, 16-bit A 16x32):
__device__ __forceinline__ int kmap(int q,int lane){ return ((q>>3)<<4)+((lane>>4)<<3)+(q&7); }

__constant__ int c_ord[6][3] = {{0,1,2},{0,2,1},{1,0,2},{1,2,0},{2,0,1},{2,1,0}};

// ---------------------------------------------------------------------------
// WMMA GEMM: C[M,N] = A[M,K] (f16) x Bw[K,N] (f16), f32 out.
// EPI: 0 store, 1 +bias, 2 +bias+gelu, 3 accumulate into C (residual).
// Block = 128 threads = 4 waves; block tile 32(M) x 64(N); each wave owns a
// 16x32 strip (two 16x16x32 WMMAs sharing one A fragment); K step 32.
// M % 32 == 0, N % 64 == 0, K % 32 == 0.
// ---------------------------------------------------------------------------
template<int EPI>
__global__ __launch_bounds__(128)
void gemm_wmma(const h16* __restrict__ A, const h16* __restrict__ Bw,
               const float* __restrict__ bias, float* __restrict__ C,
               int M, int N, int K)
{
  __shared__ h16 As[32][40];
  __shared__ h16 Bs[32][72];
  const int tid = threadIdx.x, lane = tid & 31, wave = tid >> 5;
  const int rowBase = blockIdx.y*32, colBase = blockIdx.x*64;
  const int wr = (wave>>1)*16, wc = (wave&1)*32;
  const int ldr  = tid>>2;            // 0..31
  const int ldcA = (tid&3)*8;         // A: 8 halves per thread
  const int ldcB = (tid&3)*16;        // B: 16 halves per thread (2x b128)
  v8f acc0 = {0.f,0.f,0.f,0.f,0.f,0.f,0.f,0.f};
  v8f acc1 = {0.f,0.f,0.f,0.f,0.f,0.f,0.f,0.f};
  for (int k0 = 0; k0 < K; k0 += 32) {
    const h16* gA  = &A [(size_t)(rowBase+ldr)*K + k0 + ldcA];
    const h16* gB0 = &Bw[(size_t)(k0+ldr)*N + colBase + ldcB];
#ifdef HAVE_ASYNC_LDS
    __builtin_amdgcn_global_load_async_to_lds_b128((gv4p)gA,      (lv4p)&As[ldr][ldcA],   0, 0);
    __builtin_amdgcn_global_load_async_to_lds_b128((gv4p)gB0,     (lv4p)&Bs[ldr][ldcB],   0, 0);
    __builtin_amdgcn_global_load_async_to_lds_b128((gv4p)(gB0+8), (lv4p)&Bs[ldr][ldcB+8], 0, 0);
    wait_async0();
#else
    *(uint4*)(&As[ldr][ldcA])   = *(const uint4*)(gA);
    *(uint4*)(&Bs[ldr][ldcB])   = *(const uint4*)(gB0);
    *(uint4*)(&Bs[ldr][ldcB+8]) = *(const uint4*)(gB0 + 8);
#endif
    if (k0 + 32 < K) {   // prefetch next K tile into cache (global_prefetch_b8)
      __builtin_prefetch(gA  + 32, 0, 1);
      __builtin_prefetch(gB0 + (size_t)32*N, 0, 1);
    }
    __syncthreads();
    v16h af, bf0, bf1;
#pragma unroll
    for (int q = 0; q < 16; ++q) {
      int kk = kmap(q, lane);
      af [q] = As[wr + (lane & 15)][kk];
      bf0[q] = Bs[kk][wc      + (lane & 15)];
      bf1[q] = Bs[kk][wc + 16 + (lane & 15)];
    }
    acc0 = __builtin_amdgcn_wmma_f32_16x16x32_f16(false, af, false, bf0, (short)0, acc0, false, false);
    acc1 = __builtin_amdgcn_wmma_f32_16x16x32_f16(false, af, false, bf1, (short)0, acc1, false, false);
    __syncthreads();
  }
#pragma unroll
  for (int j = 0; j < 8; ++j) {
    int m  = rowBase + wr + j + 8*(lane>>4);
    int n0 = colBase + wc + (lane & 15);
#pragma unroll
    for (int half = 0; half < 2; ++half) {
      int n = n0 + 16*half;
      float v = half ? acc1[j] : acc0[j];
      if (EPI==1 || EPI==2) v += bias[n];
      if (EPI==2) v = geluf(v);
      size_t off = (size_t)m*N + n;
      if (EPI==3) C[off] += v; else C[off] = v;
    }
  }
}

// ---------------------------------------------------------------------------
// Row LayerNorm (+ optional GELU), f32 in, f32 and/or f16 out. One block/row.
// ---------------------------------------------------------------------------
__global__ __launch_bounds__(256)
void ln_act(const float* __restrict__ in, const float* __restrict__ w, const float* __restrict__ bv,
            float* __restrict__ out32, h16* __restrict__ out16, int N, int dogelu)
{
  __shared__ float red[256];
  int r = blockIdx.x, tid = threadIdx.x;
  const float* x = in + (size_t)r*N;
  float s = 0.f, s2 = 0.f;
  for (int j = tid; j < N; j += 256){ float v = x[j]; s += v; s2 += v*v; }
  red[tid]=s; __syncthreads();
  for (int o=128;o>0;o>>=1){ if(tid<o) red[tid]+=red[tid+o]; __syncthreads(); }
  float mean = red[0]/N; __syncthreads();
  red[tid]=s2; __syncthreads();
  for (int o=128;o>0;o>>=1){ if(tid<o) red[tid]+=red[tid+o]; __syncthreads(); }
  float rstd = rsqrtf(red[0]/N - mean*mean + 1e-5f);
  for (int j = tid; j < N; j += 256){
    float v = (x[j]-mean)*rstd*w[j] + bv[j];
    if (dogelu) v = geluf(v);
    if (out32) out32[(size_t)r*N+j] = v;
    if (out16) out16[(size_t)r*N+j] = (h16)v;
  }
}

// ------------------------- weight / input conversion -----------------------
__global__ void cvt_f16(const float* __restrict__ s, h16* __restrict__ d, int n){
  int i = blockIdx.x*256 + threadIdx.x; if (i < n) d[i] = (h16)s[i];
}
__global__ void cvt_inproj(const float* __restrict__ s, h16* __restrict__ d){
  int i = blockIdx.x*256 + threadIdx.x;
  if (i >= ND_*NL_*DM_*NPAD_) return;
  int c = i % NPAD_; int rw = i / NPAD_;
  d[i] = (c < 1160) ? (h16)s[(size_t)rw*1160 + c] : (h16)0.f;
}
__global__ void build_xin(const float* __restrict__ vr, const float* __restrict__ vi, h16* __restrict__ d){
  int i = blockIdx.x*256 + threadIdx.x;
  if (i >= NTOK_*128) return;
  int c = i & 127, t = i >> 7;
  d[i] = (h16)((c < 64) ? vr[t*64 + c] : vi[t*64 + (c-64)]);
}

// ------------------- stable bitonic argsort of 2048 keys -------------------
__global__ __launch_bounds__(1024)
void sort_keys(const int* __restrict__ coords, int* __restrict__ idx, int* __restrict__ inv){
  __shared__ unsigned long long keys[2048];
  int d = blockIdx.x >> 1, b = blockIdx.x & 1, tid = threadIdx.x;
  int a0 = c_ord[d][0], a1 = c_ord[d][1], a2 = c_ord[d][2];
  for (int j = tid; j < 2048; j += 1024){
    const int* cc = coords + (size_t)(b*2048 + j)*4;
    unsigned long long key = (unsigned long long)((unsigned)cc[a0]*1048576u + (unsigned)cc[a1]*1024u + (unsigned)cc[a2]);
    keys[j] = (key << 11) | (unsigned long long)j;   // stable: tie-break by index
  }
  __syncthreads();
  for (unsigned k = 2; k <= 2048; k <<= 1){
    for (unsigned jj = k>>1; jj > 0; jj >>= 1){
      unsigned t = (unsigned)tid;
      unsigned i = ((t & ~(jj-1)) << 1) | (t & (jj-1));
      unsigned p = i | jj;
      bool up = ((i & k) == 0);
      unsigned long long a = keys[i], c2 = keys[p];
      if (up ? (a > c2) : (a < c2)) { keys[i] = c2; keys[p] = a; }
      __syncthreads();
    }
  }
  int base = (d*2 + b)*2048;
  for (int j = tid; j < 2048; j += 1024){
    int src = (int)(keys[j] & 2047ull);
    idx[base + j] = src;
    inv[base + src] = j;
  }
}

__global__ __launch_bounds__(256)
void gather_rows(const float* __restrict__ x, const int* __restrict__ idx, float* __restrict__ xd, int dsel){
  int g = blockIdx.x; int b = g >> 11, j = g & 2047;
  int src = idx[(dsel*2 + b)*2048 + j];
  xd[(size_t)g*DM_ + threadIdx.x] = x[(size_t)(b*2048 + src)*DM_ + threadIdx.x];
}
__global__ __launch_bounds__(256)
void scatter_rows(const float* __restrict__ xd, const int* __restrict__ inv, float* __restrict__ multi, int dsel){
  int g = blockIdx.x; int b = g >> 11, l = g & 2047;
  int src = inv[(dsel*2 + b)*2048 + l];
  multi[(size_t)g*(ND_*DM_) + dsel*DM_ + threadIdx.x] = xd[(size_t)(b*2048 + src)*DM_ + threadIdx.x];
}

// -------------------------- causal depthwise conv --------------------------
__global__ __launch_bounds__(256)
void conv_silu(const float* __restrict__ zx, const float* __restrict__ cw, const float* __restrict__ cb,
               float* __restrict__ xc){
  int g = blockIdx.x; int b = g >> 11, l = g & 2047;
  for (int ch = threadIdx.x; ch < CONV_; ch += 256){
    float acc = cb[ch];
#pragma unroll
    for (int k = 0; k < 4; ++k){
      int ll = l - 3 + k;
      if (ll >= 0) acc += cw[ch*4 + k] * zx[(size_t)(b*2048 + ll)*NPAD_ + 512 + ch];
    }
    xc[(size_t)g*CONV_ + ch] = siluf(acc);
  }
}

__global__ void dt_aval(const float* __restrict__ zx, const float* __restrict__ dtb,
                        const float* __restrict__ alog, float* __restrict__ dt, float* __restrict__ aval){
  int i = blockIdx.x*256 + threadIdx.x; if (i >= NTOK_*NH_) return;
  int h = i & 7; size_t r = (size_t)(i >> 3);
  float v = softplusf(zx[r*NPAD_ + 1152 + h] + dtb[h]);
  dt[i] = v;
  aval[i] = -__expf(alog[h]) * v;
}

// per-chunk inclusive cumsum of A*dt  -> Acs[(bc*8+h)*64+l], Asum[bc*8+h]
__global__ __launch_bounds__(64)
void acs_scan(const float* __restrict__ aval, float* __restrict__ acs, float* __restrict__ asum){
  __shared__ float sh[64];
  int h = blockIdx.x, bc = blockIdx.y, l = threadIdx.x;
  int b = bc >> 5, c = bc & 31;
  sh[l] = aval[(size_t)(b*2048 + c*64 + l)*8 + h];
  __syncthreads();
  for (int o = 1; o < 64; o <<= 1){
    float v = (l >= o) ? sh[l-o] : 0.f; __syncthreads();
    sh[l] += v; __syncthreads();
  }
  acs[((size_t)bc*8 + h)*64 + l] = sh[l];
  if (l == 63) asum[bc*8 + h] = sh[63];
}

__global__ __launch_bounds__(256)
void x_prep(const float* __restrict__ xc, const float* __restrict__ dt,
            h16* __restrict__ xdt, h16* __restrict__ b16, h16* __restrict__ c16){
  size_t g = blockIdx.x;
  for (int j = threadIdx.x; j < CONV_; j += 256){
    float v = xc[g*CONV_ + j];
    if (j < 512)      xdt[g*512 + j]       = (h16)(v * dt[g*8 + (j>>6)]);
    else if (j < 576) b16[g*64 + (j-512)]  = (h16)v;
    else              c16[g*64 + (j-576)]  = (h16)v;
  }
}

// ------------------ SSD pass 1: intra-chunk (diag) output ------------------
// Per (b,c): G = C·B^T (WMMA), then per head: Y_diag = (G ⊙ decay-mask) · Xdt.
__global__ __launch_bounds__(256)
void ssd_diag(const h16* __restrict__ cfull, const h16* __restrict__ bfull, const h16* __restrict__ xdt,
              const float* __restrict__ acs, float* __restrict__ Y)
{
  __shared__ h16  Cs [64][72];
  __shared__ h16  Bs2[64][72];
  __shared__ h16  Xs [64][72];
  __shared__ float Gs[64][65];
  __shared__ float Ac[8][64];
  int bc = blockIdx.x; int b = bc >> 5, c = bc & 31;
  int tid = threadIdx.x, lane = tid & 31, wave = tid >> 5;
  size_t tok0 = (size_t)b*2048 + c*64;
  for (int i = tid; i < 512; i += 256) Ac[i>>6][i&63] = acs[(size_t)bc*512 + i];
  for (int i = tid; i < 64*8; i += 256){
    int r = i >> 3, c8 = (i & 7)*8;
    *(uint4*)&Cs [r][c8] = *(const uint4*)&cfull[(tok0 + r)*64 + c8];
    *(uint4*)&Bs2[r][c8] = *(const uint4*)&bfull[(tok0 + r)*64 + c8];
  }
  __syncthreads();
  // G[l][s] = sum_n C[l][n]*B[s][n]
  for (int it = 0; it < 2; ++it){
    int t = wave + it*8; int tr = (t>>2)*16, tc = (t&3)*16;
    v8f acc = {0.f,0.f,0.f,0.f,0.f,0.f,0.f,0.f};
#pragma unroll
    for (int k0 = 0; k0 < 64; k0 += 32){
      v16h af, bf;
#pragma unroll
      for (int q = 0; q < 16; ++q){
        int kk = k0 + kmap(q, lane);
        af[q] = Cs [tr + (lane & 15)][kk];
        bf[q] = Bs2[tc + (lane & 15)][kk];
      }
      acc = __builtin_amdgcn_wmma_f32_16x16x32_f16(false, af, false, bf, (short)0, acc, false, false);
    }
#pragma unroll
    for (int j = 0; j < 8; ++j) Gs[tr + j + 8*(lane>>4)][tc + (lane & 15)] = acc[j];
  }
  __syncthreads();
  for (int h = 0; h < 8; ++h){
    for (int i = tid; i < 64*8; i += 256){
      int r = i >> 3, c8 = (i & 7)*8;
      *(uint4*)&Xs[r][c8] = *(const uint4*)&xdt[(tok0 + r)*512 + h*64 + c8];
    }
    __syncthreads();
    for (int it = 0; it < 2; ++it){
      int t = wave + it*8; int tr = (t>>2)*16, tc = (t&3)*16;
      v8f acc = {0.f,0.f,0.f,0.f,0.f,0.f,0.f,0.f};
#pragma unroll
      for (int k0 = 0; k0 < 64; k0 += 32){
        v16h af, bf;
#pragma unroll
        for (int q = 0; q < 16; ++q){
          int kk = k0 + kmap(q, lane);
          int l  = tr + (lane & 15);
          float m = (l >= kk) ? Gs[l][kk]*__expf(Ac[h][l] - Ac[h][kk]) : 0.f;
          af[q] = (h16)m;
          bf[q] = Xs[kk][tc + (lane & 15)];
        }
        acc = __builtin_amdgcn_wmma_f32_16x16x32_f16(false, af, false, bf, (short)0, acc, false, false);
      }
#pragma unroll
      for (int j = 0; j < 8; ++j){
        int l = tr + j + 8*(lane>>4);
        Y[(tok0 + l)*512 + h*64 + tc + (lane & 15)] = acc[j];
      }
    }
    __syncthreads();
  }
}

// ------------- SSD pass 2: per-chunk state  S[p][n] = X'^T · B --------------
__global__ __launch_bounds__(128)
void ssd_states(const h16* __restrict__ bfull, const h16* __restrict__ xdt,
                const float* __restrict__ acs, float* __restrict__ states)
{
  __shared__ h16 Xd [64][72];
  __shared__ h16 Bs2[64][72];
  __shared__ float Ac[64];
  int h = blockIdx.x, bc = blockIdx.y; int b = bc >> 5, c = bc & 31;
  int tid = threadIdx.x, lane = tid & 31, wave = tid >> 5;
  size_t tok0 = (size_t)b*2048 + c*64;
  if (tid < 64) Ac[tid] = acs[((size_t)bc*8 + h)*64 + tid];
  for (int i = tid; i < 64*8; i += 128){
    int r = i >> 3, c8 = (i & 7)*8;
    *(uint4*)&Bs2[r][c8] = *(const uint4*)&bfull[(tok0 + r)*64 + c8];
  }
  __syncthreads();
  float asum = Ac[63];
  for (int i = tid; i < 4096; i += 128){
    int l = i >> 6, p = i & 63;
    Xd[l][p] = (h16)((float)xdt[(tok0 + l)*512 + h*64 + p] * __expf(asum - Ac[l]));
  }
  __syncthreads();
  size_t sbase = ((size_t)bc*8 + h)*4096;
  for (int it = 0; it < 4; ++it){
    int t = wave*4 + it; int tr = (t>>2)*16, tc = (t&3)*16;
    v8f acc = {0.f,0.f,0.f,0.f,0.f,0.f,0.f,0.f};
#pragma unroll
    for (int k0 = 0; k0 < 64; k0 += 32){
      v16h af, bf;
#pragma unroll
      for (int q = 0; q < 16; ++q){
        int kk = k0 + kmap(q, lane);
        af[q] = Xd [kk][tr + (lane & 15)];
        bf[q] = Bs2[kk][tc + (lane & 15)];
      }
      acc = __builtin_amdgcn_wmma_f32_16x16x32_f16(false, af, false, bf, (short)0, acc, false, false);
    }
#pragma unroll
    for (int j = 0; j < 8; ++j){
      int p = tr + j + 8*(lane>>4), n = tc + (lane & 15);
      states[sbase + p*64 + n] = acc[j];
    }
  }
}

// ------- SSD pass 3: sequential inter-chunk scan (register resident) -------
// states[c] := prefix S before chunk c;  S = S*exp(Asum_c) + states_c.
__global__ __launch_bounds__(256)
void ssd_scan(float* __restrict__ states, const float* __restrict__ asum){
  int bh = blockIdx.x; int b = bh >> 3, h = bh & 7;
  int tid = threadIdx.x;
  float S[16];
#pragma unroll
  for (int r = 0; r < 16; ++r) S[r] = 0.f;
  for (int c = 0; c < 32; ++c){
    size_t base = ((size_t)(b*32 + c)*8 + h)*4096;
    float e = __expf(asum[(b*32 + c)*8 + h]);
#pragma unroll
    for (int r = 0; r < 16; ++r){
      size_t o = base + tid + r*256;
      float st = states[o];
      states[o] = S[r];
      S[r] = S[r]*e + st;
    }
  }
}

// ------------- SSD pass 4: Y += exp(Acs[l]) * (C · S_prev^T) ---------------
__global__ __launch_bounds__(128)
void ssd_yoff(const h16* __restrict__ cfull, const float* __restrict__ states,
              const float* __restrict__ acs, float* __restrict__ Y)
{
  __shared__ h16 Cs[64][72];
  __shared__ h16 Sp[64][72];
  __shared__ float Ac[64];
  int h = blockIdx.x, bc = blockIdx.y; int b = bc >> 5, c = bc & 31;
  int tid = threadIdx.x, lane = tid & 31, wave = tid >> 5;
  size_t tok0 = (size_t)b*2048 + c*64;
  if (tid < 64) Ac[tid] = acs[((size_t)bc*8 + h)*64 + tid];
  for (int i = tid; i < 64*8; i += 128){
    int r = i >> 3, c8 = (i & 7)*8;
    *(uint4*)&Cs[r][c8] = *(const uint4*)&cfull[(tok0 + r)*64 + c8];
  }
  size_t sbase = ((size_t)bc*8 + h)*4096;
  for (int i = tid; i < 4096; i += 128) Sp[i>>6][i&63] = (h16)states[sbase + i];
  __syncthreads();
  for (int it = 0; it < 4; ++it){
    int t = wave*4 + it; int tr = (t>>2)*16, tc = (t&3)*16;
    v8f acc = {0.f,0.f,0.f,0.f,0.f,0.f,0.f,0.f};
#pragma unroll
    for (int k0 = 0; k0 < 64; k0 += 32){
      v16h af, bf;
#pragma unroll
      for (int q = 0; q < 16; ++q){
        int kk = k0 + kmap(q, lane);
        af[q] = Cs[tr + (lane & 15)][kk];
        bf[q] = Sp[tc + (lane & 15)][kk];
      }
      acc = __builtin_amdgcn_wmma_f32_16x16x32_f16(false, af, false, bf, (short)0, acc, false, false);
    }
#pragma unroll
    for (int j = 0; j < 8; ++j){
      int l = tr + j + 8*(lane>>4);
      Y[(tok0 + l)*512 + h*64 + tc + (lane & 15)] += __expf(Ac[l]) * acc[j];
    }
  }
}

// ------------------- gating, D skip, RMS norm -> f16 -----------------------
__global__ __launch_bounds__(256)
void gate_rms(const float* __restrict__ Y, const float* __restrict__ xc, const float* __restrict__ zxb,
              const float* __restrict__ Dp, const float* __restrict__ rw, h16* __restrict__ y16)
{
  __shared__ float red[256];
  size_t g = blockIdx.x; int tid = threadIdx.x;
  float v[2]; float s2 = 0.f;
#pragma unroll
  for (int r = 0; r < 2; ++r){
    int di = tid + r*256;
    float yv = Y[g*512 + di] + Dp[di>>6]*xc[g*CONV_ + di];
    float z  = zxb[g*NPAD_ + di];
    yv *= siluf(z);
    v[r] = yv; s2 += yv*yv;
  }
  red[tid] = s2; __syncthreads();
  for (int o = 128; o > 0; o >>= 1){ if (tid < o) red[tid] += red[tid+o]; __syncthreads(); }
  float scale = rsqrtf(red[0]/512.f + 1e-5f);
#pragma unroll
  for (int r = 0; r < 2; ++r){
    int di = tid + r*256;
    y16[g*512 + di] = (h16)(v[r]*scale*rw[di]);
  }
}

// ----------------------- pooling (p1==p2==p4==mean) ------------------------
__global__ __launch_bounds__(256)
void mean_pool(const float* __restrict__ fused, float* __restrict__ mp){
  int b = blockIdx.x, f = threadIdx.x;
  float s = 0.f;
  for (int l = 0; l < 2048; ++l) s += fused[(size_t)(b*2048 + l)*DM_ + f];
  mp[b*DM_ + f] = s*(1.0f/2048.0f);
}

// ----------------------------- classifier head -----------------------------
__global__ __launch_bounds__(256)
void classifier(const float* __restrict__ mp,
                const float* __restrict__ lnw, const float* __restrict__ lnb,
                const float* __restrict__ w1, const float* __restrict__ b1,
                const float* __restrict__ w2, const float* __restrict__ b2,
                const float* __restrict__ w3, const float* __restrict__ b3,
                float* __restrict__ out)
{
  __shared__ float pl[768];
  __shared__ float h1[512];
  __shared__ float h2[256];
  __shared__ float red[256];
  int tid = threadIdx.x;
  for (int b = 0; b < 2; ++b){
    for (int j = tid; j < 768; j += 256) pl[j] = mp[b*256 + (j & 255)];
    __syncthreads();
    float s = 0.f, s2 = 0.f;
    for (int j = tid; j < 768; j += 256){ float x = pl[j]; s += x; s2 += x*x; }
    red[tid] = s; __syncthreads();
    for (int o = 128; o > 0; o >>= 1){ if (tid < o) red[tid] += red[tid+o]; __syncthreads(); }
    float mean = red[0]/768.f; __syncthreads();
    red[tid] = s2; __syncthreads();
    for (int o = 128; o > 0; o >>= 1){ if (tid < o) red[tid] += red[tid+o]; __syncthreads(); }
    float rstd = rsqrtf(red[0]/768.f - mean*mean + 1e-5f);
    __syncthreads();
    for (int j = tid; j < 768; j += 256) pl[j] = (pl[j]-mean)*rstd*lnw[j] + lnb[j];
    __syncthreads();
    for (int o = tid; o < 512; o += 256){
      float a = b1[o];
      for (int j = 0; j < 768; ++j) a += pl[j]*w1[j*512 + o];
      h1[o] = geluf(a);
    }
    __syncthreads();
    for (int o = tid; o < 256; o += 256){
      float a = b2[o];
      for (int j = 0; j < 512; ++j) a += h1[j]*w2[j*256 + o];
      h2[o] = geluf(a);
    }
    __syncthreads();
    if (tid < NCLS_){
      float a = b3[tid];
      for (int j = 0; j < 256; ++j) a += h2[j]*w3[j*NCLS_ + tid];
      out[b*NCLS_ + tid] = a;
    }
    __syncthreads();
  }
}

// ---------------------------------------------------------------------------
extern "C" void kernel_launch(void* const* d_in, const int* in_sizes, int n_in,
                              void* d_out, int out_size, void* d_ws, size_t ws_size,
                              hipStream_t stream)
{
  (void)in_sizes; (void)n_in; (void)out_size; (void)ws_size;
  const float* v_real    = (const float*)d_in[0];
  const float* v_imag    = (const float*)d_in[1];
  const int*   coords    = (const int*)  d_in[2];
  const float* w_in      = (const float*)d_in[3];
  const float* b_in      = (const float*)d_in[4];
  const float* ln_in_w   = (const float*)d_in[5];
  const float* ln_in_b   = (const float*)d_in[6];
  const float* blk_ln_w  = (const float*)d_in[7];
  const float* blk_ln_b  = (const float*)d_in[8];
  const float* in_proj_w = (const float*)d_in[9];
  const float* conv_w    = (const float*)d_in[10];
  const float* conv_b    = (const float*)d_in[11];
  const float* dt_bias   = (const float*)d_in[12];
  const float* A_log     = (const float*)d_in[13];
  const float* D_param   = (const float*)d_in[14];
  const float* rms_w     = (const float*)d_in[15];
  const float* out_proj_w= (const float*)d_in[16];
  const float* fus_ln_w  = (const float*)d_in[17];
  const float* fus_ln_b  = (const float*)d_in[18];
  const float* w_fuse    = (const float*)d_in[19];
  const float* b_fuse    = (const float*)d_in[20];
  const float* cls_ln_w  = (const float*)d_in[21];
  const float* cls_ln_b  = (const float*)d_in[22];
  const float* w1 = (const float*)d_in[23];
  const float* b1 = (const float*)d_in[24];
  const float* w2 = (const float*)d_in[25];
  const float* b2 = (const float*)d_in[26];
  const float* w3 = (const float*)d_in[27];
  const float* b3 = (const float*)d_in[28];

  // ------------- workspace carve-out (~106 MB, 256B aligned) ---------------
  char* base = (char*)d_ws; size_t off = 0;
  auto alloc = [&](size_t bytes)->char*{
    char* p = base + off; off = (off + bytes + 255) & ~(size_t)255; return p;
  };
  h16*  w16_in     = (h16*) alloc((size_t)128*256*2);
  h16*  w16_inproj = (h16*) alloc((size_t)12*256*NPAD_*2);
  h16*  w16_outproj= (h16*) alloc((size_t)12*512*256*2);
  h16*  w16_fuse   = (h16*) alloc((size_t)1536*256*2);
  h16*  xin16      = (h16*) alloc((size_t)NTOK_*128*2);
  float* x32       = (float*)alloc((size_t)NTOK_*256*4);
  int*  idxb       = (int*)  alloc((size_t)12*2048*4);
  int*  invb       = (int*)  alloc((size_t)12*2048*4);
  float* xd32      = (float*)alloc((size_t)NTOK_*256*4);
  h16*  h16buf     = (h16*) alloc((size_t)NTOK_*256*2);
  float* zx        = (float*)alloc((size_t)NTOK_*NPAD_*4);
  float* xconv     = (float*)alloc((size_t)NTOK_*CONV_*4);
  float* dtbuf     = (float*)alloc((size_t)NTOK_*8*4);
  float* avalb     = (float*)alloc((size_t)NTOK_*8*4);
  float* acsb      = (float*)alloc((size_t)NTOK_*8*4);
  float* asumb     = (float*)alloc((size_t)2*32*8*4);
  h16*  xdt16      = (h16*) alloc((size_t)NTOK_*512*2);
  h16*  b16        = (h16*) alloc((size_t)NTOK_*64*2);
  h16*  c16        = (h16*) alloc((size_t)NTOK_*64*2);
  float* Y32       = (float*)alloc((size_t)NTOK_*512*4);
  float* st32      = (float*)alloc((size_t)2*32*8*4096*4);
  h16*  y16        = (h16*) alloc((size_t)NTOK_*512*2);
  float* multi32   = (float*)alloc((size_t)NTOK_*1536*4);
  float* mpool     = (float*)alloc((size_t)2*256*4);
  // time-disjoint aliases
  float* embtmp  = Y32;                 // used only before the direction loop
  h16*  multi16  = (h16*)zx;            // zx dead after directions (12.6MB<=19.9MB)
  float* fused32 = xconv;               // xconv dead after directions (4MB<=10.5MB)

  // ------------------------- weight conversion -----------------------------
  int n;
  n = 128*256;        cvt_f16   <<<(n+255)/256,256,0,stream>>>(w_in, w16_in, n);
  n = 12*256*NPAD_;   cvt_inproj<<<(n+255)/256,256,0,stream>>>(in_proj_w, w16_inproj);
  n = 12*512*256;     cvt_f16   <<<(n+255)/256,256,0,stream>>>(out_proj_w, w16_outproj, n);
  n = 1536*256;       cvt_f16   <<<(n+255)/256,256,0,stream>>>(w_fuse, w16_fuse, n);
  n = NTOK_*128;      build_xin <<<(n+255)/256,256,0,stream>>>(v_real, v_imag, xin16);
  sort_keys<<<12,1024,0,stream>>>(coords, idxb, invb);

  // ------------------------------ embed ------------------------------------
  gemm_wmma<1><<<dim3(256/64, NTOK_/32),128,0,stream>>>(xin16, w16_in, b_in, embtmp, NTOK_, 256, 128);
  ln_act<<<NTOK_,256,0,stream>>>(embtmp, ln_in_w, ln_in_b, x32, (h16*)nullptr, 256, 1);

  // ------------------------- 6 directions × 2 blocks -----------------------
  for (int d = 0; d < ND_; ++d){
    gather_rows<<<NTOK_,256,0,stream>>>(x32, idxb, xd32, d);
    for (int i = 0; i < NL_; ++i){
      int li = d*2 + i;
      ln_act<<<NTOK_,256,0,stream>>>(xd32, blk_ln_w + li*256, blk_ln_b + li*256,
                                     (float*)nullptr, h16buf, 256, 0);
      gemm_wmma<0><<<dim3(NPAD_/64, NTOK_/32),128,0,stream>>>(h16buf, w16_inproj + (size_t)li*256*NPAD_,
                                                              (const float*)nullptr, zx, NTOK_, NPAD_, 256);
      conv_silu<<<NTOK_,256,0,stream>>>(zx, conv_w + (size_t)li*CONV_*4, conv_b + (size_t)li*CONV_, xconv);
      n = NTOK_*8;
      dt_aval<<<(n+255)/256,256,0,stream>>>(zx, dt_bias + li*8, A_log + li*8, dtbuf, avalb);
      acs_scan<<<dim3(8,64),64,0,stream>>>(avalb, acsb, asumb);
      x_prep<<<NTOK_,256,0,stream>>>(xconv, dtbuf, xdt16, b16, c16);
      ssd_diag  <<<64,256,0,stream>>>(c16, b16, xdt16, acsb, Y32);
      ssd_states<<<dim3(8,64),128,0,stream>>>(b16, xdt16, acsb, st32);
      ssd_scan  <<<16,256,0,stream>>>(st32, asumb);
      ssd_yoff  <<<dim3(8,64),128,0,stream>>>(c16, st32, acsb, Y32);
      gate_rms<<<NTOK_,256,0,stream>>>(Y32, xconv, zx, D_param + li*8, rms_w + (size_t)li*512, y16);
      gemm_wmma<3><<<dim3(256/64, NTOK_/32),128,0,stream>>>(y16, w16_outproj + (size_t)li*512*256,
                                                            (const float*)nullptr, xd32, NTOK_, 256, 512);
    }
    scatter_rows<<<NTOK_,256,0,stream>>>(xd32, invb, multi32, d);
  }

  // --------------------------- fuse + head ---------------------------------
  ln_act<<<NTOK_,256,0,stream>>>(multi32, fus_ln_w, fus_ln_b, (float*)nullptr, multi16, 1536, 0);
  gemm_wmma<2><<<dim3(256/64, NTOK_/32),128,0,stream>>>(multi16, w16_fuse, b_fuse, fused32, NTOK_, 256, 1536);
  mean_pool<<<2,256,0,stream>>>(fused32, mpool);
  classifier<<<1,256,0,stream>>>(mpool, cls_ln_w, cls_ln_b, w1, b1, w2, b2, w3, b3, (float*)d_out);
}
